// OrthoNoChunkPhasor_37082747633905
// MI455X (gfx1250) — matrix operand order repaired and checked by reading, not compiled, not run on hardware
//
#include <hip/hip_runtime.h>
#include <hip/hip_bf16.h>

// ---------------------------------------------------------------------------
// Types for CDNA5 WMMA (wave32, v_wmma_f32_16x16x32_bf16)
// ---------------------------------------------------------------------------
typedef __bf16 bf16_t;
typedef __attribute__((ext_vector_type(16))) __bf16 v16bf;
typedef __attribute__((ext_vector_type(8)))  __bf16 v8bf;
typedef __attribute__((ext_vector_type(8)))  float  v8f;
typedef __attribute__((ext_vector_type(4)))  int    v4i;

typedef __attribute__((address_space(1))) v4i gas_v4i;  // global
typedef __attribute__((address_space(3))) v4i las_v4i;  // LDS

// Problem constants (match reference)
#define DIMD   1024
#define SEQ    2048
#define BATCH  4
#define MTOT   (BATCH * SEQ)          // 8192 rows
#define TELEMS ((size_t)MTOT * DIMD)  // 8,388,608
#define LN_EPS 1e-5f
#define INV_SQRT_D 0.03125f           // 1/sqrt(1024)

// Scan blocking
#define SCH 16                        // chunks along SEQ
#define SCL (SEQ / SCH)               // 128 steps per chunk

static __device__ __forceinline__ bf16_t f2bf(float f) {
  unsigned u = __builtin_bit_cast(unsigned, f);
  unsigned r = u + 0x7FFFu + ((u >> 16) & 1u);   // round-to-nearest-even
  unsigned short h = (unsigned short)(r >> 16);
  return __builtin_bit_cast(bf16_t, h);
}

// ---------------------------------------------------------------------------
// CDNA5 async global->LDS copy (ASYNCcnt path, no VGPR data staging)
// ---------------------------------------------------------------------------
static __device__ __forceinline__ void async_b128(const bf16_t* g, bf16_t* l) {
#if __has_builtin(__builtin_amdgcn_global_load_async_to_lds_b128)
  __builtin_amdgcn_global_load_async_to_lds_b128(
      (gas_v4i*)const_cast<bf16_t*>(g),
      (las_v4i*)l, 0, 0);
#else
  asm volatile("global_load_async_to_lds_b128 %0, %1, off"
               :: "v"((unsigned)(unsigned long long)
                        (__attribute__((address_space(3))) bf16_t*)l),
                  "v"((unsigned long long)g)
               : "memory");
#endif
}

static __device__ __forceinline__ void wait_async0() {
#if __has_builtin(__builtin_amdgcn_s_wait_asynccnt)
  __builtin_amdgcn_s_wait_asynccnt(0);
#else
  asm volatile("s_wait_asynccnt 0x0" ::: "memory");
#endif
}

// ---------------------------------------------------------------------------
// fp32 -> bf16 conversion (grid-stride)
// ---------------------------------------------------------------------------
__global__ __launch_bounds__(256) void cvt_f32_bf16(const float* __restrict__ in,
                                                    bf16_t* __restrict__ out, int n) {
  for (int i = blockIdx.x * blockDim.x + threadIdx.x; i < n;
       i += gridDim.x * blockDim.x) {
    out[i] = f2bf(in[i]);
  }
}

// ---------------------------------------------------------------------------
// WMMA GEMM:  C[M,N](f32) = A[M,K](bf16) x W[N,K](bf16)^T + bias (+ residual)
// Block tile 128x128, 8 waves (wave32), wave tile 32x64 = 2x4 WMMA accums.
// K staged in double-buffered LDS via GLOBAL_LOAD_ASYNC_TO_LDS_B128; one
// barrier per K-step; s_wait_asynccnt gates tile arrival.
// ---------------------------------------------------------------------------
#define BM 128
#define BN 128
#define BK 32
#define LDK 40   // padded row stride (bf16 elems) to spread LDS banks

__global__ __launch_bounds__(256) void wmma_gemm_bias(
    const bf16_t* __restrict__ A,      // M x K row-major
    const bf16_t* __restrict__ W,      // N x K row-major (weights)
    const float*  __restrict__ bias,   // N
    const float*  __restrict__ residual, // M x N or nullptr
    float* __restrict__ C,             // M x N
    int M, int N, int K)
{
  __shared__ alignas(16) bf16_t As[2][BM * LDK];
  __shared__ alignas(16) bf16_t Bs[2][BN * LDK];

  const int tid   = threadIdx.x;
  const int lane  = tid & 31;
  const int wave  = tid >> 5;
  const int lm    = lane & 15;         // matrix row/col within 16
  const int hi    = lane >> 4;         // half-wave selector
  const int waveM = wave & 3;          // 0..3 -> 32-row slab
  const int waveN = wave >> 2;         // 0..1 -> 64-col slab
  const int blockM = blockIdx.y * BM;
  const int blockN = blockIdx.x * BN;

  // Async staging: 4 threads per 32-elem row, 16B (8 bf16) per thread.
  const int chunk = tid & 3;           // 0..3 -> 16B chunk in row
  const int row0  = tid >> 2;          // 0..63

  v8f acc[2][4];
#pragma unroll
  for (int mi = 0; mi < 2; ++mi)
#pragma unroll
    for (int ni = 0; ni < 4; ++ni)
#pragma unroll
      for (int e = 0; e < 8; ++e) acc[mi][ni][e] = 0.0f;

  auto issue_tile = [&](int kk, int buf) {
    const bf16_t* gA = A + (size_t)blockM * K + kk;
    const bf16_t* gB = W + (size_t)blockN * K + kk;
#pragma unroll
    for (int j = 0; j < 2; ++j) {
      const int row = row0 + j * 64;
      async_b128(gA + (size_t)row * K + chunk * 8, &As[buf][row * LDK + chunk * 8]);
      async_b128(gB + (size_t)row * K + chunk * 8, &Bs[buf][row * LDK + chunk * 8]);
    }
  };

  issue_tile(0, 0);
  int buf = 0;

  for (int kk = 0; kk < K; kk += BK) {
    wait_async0();       // own wave's async tile writes for `buf` landed
    __syncthreads();     // all waves' tiles landed; buf^1 free for reuse

    if (kk + BK < K)     // overlap next tile's DMA with this tile's WMMAs
      issue_tile(kk + BK, buf ^ 1);

    // A fragment, 16x32 bf16 (ISA layout): lane<16 K={0..7,16..23}, lane>=16 +8
    v16bf af[2];
#pragma unroll
    for (int mi = 0; mi < 2; ++mi) {
      const bf16_t* p = &As[buf][(waveM * 32 + mi * 16 + lm) * LDK + hi * 8];
      v8bf lo  = *(const v8bf*)(p);
      v8bf hi8 = *(const v8bf*)(p + 16);
      af[mi] = __builtin_shufflevector(lo, hi8, 0,1,2,3,4,5,6,7,8,9,10,11,12,13,14,15);
    }
    // B fragment, 32x16 bf16: lane n=lm, K={0..7,16..23} (+8 for lane>=16), matching A
    v16bf bfm[4];
#pragma unroll
    for (int ni = 0; ni < 4; ++ni) {
      const bf16_t* p = &Bs[buf][(waveN * 64 + ni * 16 + lm) * LDK + hi * 8];
      v8bf lo = *(const v8bf*)(p);
      v8bf h8 = *(const v8bf*)(p + 16);
      bfm[ni] = __builtin_shufflevector(lo, h8, 0,1,2,3,4,5,6,7,8,9,10,11,12,13,14,15);
    }

#pragma unroll
    for (int mi = 0; mi < 2; ++mi)
#pragma unroll
      for (int ni = 0; ni < 4; ++ni)
        acc[mi][ni] = __builtin_amdgcn_wmma_f32_16x16x32_bf16(
            false, af[mi], false, bfm[ni], (short)0, acc[mi][ni], false, false);

    buf ^= 1;
  }

  // Epilogue: C VGPR r -> row = r + 8*hi (ISA C/D layout), col = lm
#pragma unroll
  for (int mi = 0; mi < 2; ++mi) {
    const int rowBase = blockM + waveM * 32 + mi * 16 + hi * 8;
#pragma unroll
    for (int ni = 0; ni < 4; ++ni) {
      const int col = blockN + waveN * 64 + ni * 16 + lm;
      const float bv = bias[col];
#pragma unroll
      for (int r = 0; r < 8; ++r) {
        const size_t idx = (size_t)(rowBase + r) * N + col;
        float v = acc[mi][ni][r] + bv;
        if (residual) v += residual[idx];
        C[idx] = v;
      }
    }
  }
}

// ---------------------------------------------------------------------------
// Phase kernel: kp = base + km*ms ; qp = base + qm*ms
//   vc = value*cos(kp), vs = value*sin(kp), cq = cos(qp), sq = sin(qp)
// cq/sq may alias km/qm (pure elementwise, same index) -> no __restrict__.
// ---------------------------------------------------------------------------
__global__ __launch_bounds__(256) void phase_kernel(
    const float* __restrict__ base,   // SEQ*DIMD
    const float* km, const float* qm,
    const float* __restrict__ value,
    const float* __restrict__ mod_scale,
    float* vc, float* vs, float* cqo, float* sqo)
{
  const float ms = mod_scale[0];
  const size_t total = TELEMS;
  const size_t mask = (size_t)SEQ * DIMD - 1;   // S*D is a power of two
  for (size_t i = blockIdx.x * blockDim.x + threadIdx.x; i < total;
       i += (size_t)gridDim.x * blockDim.x) {
    const float bp = base[i & mask];
    const float kp = bp + km[i] * ms;
    const float qp = bp + qm[i] * ms;
    float sk, ck, sq2, cq2;
    __sincosf(kp, &sk, &ck);
    __sincosf(qp, &sq2, &cq2);
    const float v = value[i];
    vc[i]  = v * ck;
    vs[i]  = v * sk;
    cqo[i] = cq2;
    sqo[i] = sq2;
  }
}

// ---------------------------------------------------------------------------
// Blocked scan over SEQ, 16 chunks of 128. One thread per (b, d) channel
// per chunk; coalesced across d.
// ---------------------------------------------------------------------------
__global__ __launch_bounds__(256) void scan_partial(
    const float* __restrict__ vc, const float* __restrict__ vs,
    float* __restrict__ sumR, float* __restrict__ sumI)
{
  const int d  = blockIdx.x * blockDim.x + threadIdx.x;  // 0..1023
  const int ch = blockIdx.y;                             // 0..15
  const int b  = blockIdx.z;                             // 0..3
  size_t base = ((size_t)b * SEQ + (size_t)ch * SCL) * DIMD + d;
  float aR = 0.0f, aI = 0.0f;
  for (int s = 0; s < SCL; ++s) {
    aR += vc[base]; aI += vs[base];
    base += DIMD;
  }
  const size_t o = ((size_t)b * SCH + ch) * DIMD + d;
  sumR[o] = aR; sumI[o] = aI;
}

__global__ __launch_bounds__(256) void scan_offsets(float* sumR, float* sumI)
{
  const int idx = blockIdx.x * blockDim.x + threadIdx.x;  // 0..4095 = b*D+d
  const int b = idx >> 10, d = idx & (DIMD - 1);
  float aR = 0.0f, aI = 0.0f;
  for (int ch = 0; ch < SCH; ++ch) {
    const size_t j = ((size_t)b * SCH + ch) * DIMD + d;
    const float tR = sumR[j], tI = sumI[j];
    sumR[j] = aR; sumI[j] = aI;   // exclusive prefix in-place
    aR += tR; aI += tI;
  }
}

__global__ __launch_bounds__(256) void scan_apply(
    const float* __restrict__ vc, const float* __restrict__ vs,
    const float* __restrict__ sumR, const float* __restrict__ sumI,
    const float* __restrict__ cq, const float* __restrict__ sq,
    float* __restrict__ retr)
{
  const int d  = blockIdx.x * blockDim.x + threadIdx.x;
  const int ch = blockIdx.y;
  const int b  = blockIdx.z;
  const size_t o = ((size_t)b * SCH + ch) * DIMD + d;
  float aR = sumR[o], aI = sumI[o];
  size_t base = ((size_t)b * SEQ + (size_t)ch * SCL) * DIMD + d;
  for (int s = 0; s < SCL; ++s) {
    aR += vc[base]; aI += vs[base];
    retr[base] = (aR * cq[base] + aI * sq[base]) * INV_SQRT_D;
    base += DIMD;
  }
}

// ---------------------------------------------------------------------------
// LayerNorm over last dim (1024), one 256-thread block per row; bf16 out.
// ---------------------------------------------------------------------------
__global__ __launch_bounds__(256) void layernorm_bf16(
    const float* __restrict__ in, const float* __restrict__ g,
    const float* __restrict__ bta, bf16_t* __restrict__ out)
{
  const int row = blockIdx.x;
  const int tid = threadIdx.x;
  const float* r = in + (size_t)row * DIMD;

  float s = 0.0f, s2 = 0.0f;
#pragma unroll
  for (int i = tid; i < DIMD; i += 256) {
    const float v = r[i];
    s += v; s2 += v * v;
  }
  __shared__ float red0[256];
  __shared__ float red1[256];
  red0[tid] = s; red1[tid] = s2;
  __syncthreads();
  for (int off = 128; off > 0; off >>= 1) {
    if (tid < off) { red0[tid] += red0[tid + off]; red1[tid] += red1[tid + off]; }
    __syncthreads();
  }
  const float mu  = red0[0] * (1.0f / DIMD);
  const float var = red1[0] * (1.0f / DIMD) - mu * mu;
  const float rinv = rsqrtf(var + LN_EPS);
#pragma unroll
  for (int i = tid; i < DIMD; i += 256) {
    out[(size_t)row * DIMD + i] = f2bf((r[i] - mu) * rinv * g[i] + bta[i]);
  }
}

// ---------------------------------------------------------------------------
// Host orchestration
// ---------------------------------------------------------------------------
extern "C" void kernel_launch(void* const* d_in, const int* in_sizes, int n_in,
                              void* d_out, int out_size, void* d_ws, size_t ws_size,
                              hipStream_t stream) {
  (void)in_sizes; (void)n_in; (void)out_size; (void)ws_size;

  const float* x    = (const float*)d_in[0];
  const float* base = (const float*)d_in[1];
  const float* Wk   = (const float*)d_in[2];  const float* bk = (const float*)d_in[3];
  const float* Wv   = (const float*)d_in[4];  const float* bv = (const float*)d_in[5];
  const float* Wq   = (const float*)d_in[6];  const float* bq = (const float*)d_in[7];
  const float* Wkm  = (const float*)d_in[8];  const float* bkm = (const float*)d_in[9];
  const float* Wqm  = (const float*)d_in[10]; const float* bqm = (const float*)d_in[11];
  const float* ms   = (const float*)d_in[12];
  const float* ln_g = (const float*)d_in[13]; const float* ln_b = (const float*)d_in[14];
  const float* Wo   = (const float*)d_in[15]; const float* bo = (const float*)d_in[16];
  float* out = (float*)d_out;

  // Workspace carve-up (aggressively reused)
  char* ws = (char*)d_ws;
  size_t off = 0;
  auto carve = [&](size_t bytes) -> char* {
    char* p = ws + off;
    off += (bytes + 255) & ~(size_t)255;
    return p;
  };
  const size_t F32 = TELEMS * sizeof(float);     // 32 MB token-major f32
  const size_t BF  = TELEMS * sizeof(bf16_t);    // 16 MB token-major bf16
  const size_t WB  = (size_t)DIMD * DIMD * sizeof(bf16_t); // 2 MB weight bf16

  bf16_t* xb   = (bf16_t*)carve(BF);
  bf16_t* Wkb  = (bf16_t*)carve(WB);
  bf16_t* Wvb  = (bf16_t*)carve(WB);
  bf16_t* Wqb  = (bf16_t*)carve(WB);
  bf16_t* Wkmb = (bf16_t*)carve(WB);
  bf16_t* Wqmb = (bf16_t*)carve(WB);
  bf16_t* Wob  = (bf16_t*)carve(WB);
  float*  keyB  = (float*)carve(F32);   // key  -> later vc
  float*  valB  = (float*)carve(F32);   // value-> later retrieved
  float*  qryB  = (float*)carve(F32);   // query-> later vs
  bf16_t* keyb  = (bf16_t*)carve(BF);   // bf16 key -> later normed(bf16)
  bf16_t* qryb  = (bf16_t*)carve(BF);
  float*  kmB   = (float*)carve(F32);   // km -> cq (in place)
  float*  qmB   = (float*)carve(F32);   // qm -> sq (in place)
  float*  sumR  = (float*)carve((size_t)BATCH * SCH * DIMD * sizeof(float));
  float*  sumI  = (float*)carve((size_t)BATCH * SCH * DIMD * sizeof(float));

  const int NW = DIMD * DIMD;   // 1M weight elems
  const dim3 gemmGrid(DIMD / BN, MTOT / BM);   // (8, 64)

  // 1) fp32 -> bf16 conversions
  cvt_f32_bf16<<<2048, 256, 0, stream>>>(x,   xb,   (int)TELEMS);
  cvt_f32_bf16<<<1024, 256, 0, stream>>>(Wk,  Wkb,  NW);
  cvt_f32_bf16<<<1024, 256, 0, stream>>>(Wv,  Wvb,  NW);
  cvt_f32_bf16<<<1024, 256, 0, stream>>>(Wq,  Wqb,  NW);
  cvt_f32_bf16<<<1024, 256, 0, stream>>>(Wkm, Wkmb, NW);
  cvt_f32_bf16<<<1024, 256, 0, stream>>>(Wqm, Wqmb, NW);
  cvt_f32_bf16<<<1024, 256, 0, stream>>>(Wo,  Wob,  NW);

  // 2) K / V / Q projections (WMMA bf16, f32 accumulate)
  wmma_gemm_bias<<<gemmGrid, 256, 0, stream>>>(xb, Wkb, bk, nullptr, keyB, MTOT, DIMD, DIMD);
  wmma_gemm_bias<<<gemmGrid, 256, 0, stream>>>(xb, Wvb, bv, nullptr, valB, MTOT, DIMD, DIMD);
  wmma_gemm_bias<<<gemmGrid, 256, 0, stream>>>(xb, Wqb, bq, nullptr, qryB, MTOT, DIMD, DIMD);

  // 3) demote key/query for the modulator GEMMs
  cvt_f32_bf16<<<2048, 256, 0, stream>>>(keyB, keyb, (int)TELEMS);
  cvt_f32_bf16<<<2048, 256, 0, stream>>>(qryB, qryb, (int)TELEMS);

  // 4) phase modulators
  wmma_gemm_bias<<<gemmGrid, 256, 0, stream>>>(keyb, Wkmb, bkm, nullptr, kmB, MTOT, DIMD, DIMD);
  wmma_gemm_bias<<<gemmGrid, 256, 0, stream>>>(qryb, Wqmb, bqm, nullptr, qmB, MTOT, DIMD, DIMD);

  // 5) fused sincos / phase products.  vc->keyB, vs->qryB, cq/sq in place.
  phase_kernel<<<4096, 256, 0, stream>>>(base, kmB, qmB, valB, ms,
                                         keyB, qryB, kmB, qmB);

  // 6) blocked cumsum + fused retrieval (retrieved -> valB)
  const dim3 scanGrid(DIMD / 256, SCH, BATCH);
  scan_partial<<<scanGrid, 256, 0, stream>>>(keyB, qryB, sumR, sumI);
  scan_offsets<<<16, 256, 0, stream>>>(sumR, sumI);
  scan_apply<<<scanGrid, 256, 0, stream>>>(keyB, qryB, sumR, sumI, kmB, qmB, valB);

  // 7) LayerNorm -> bf16 (reuse keyb as normed buffer)
  layernorm_bf16<<<MTOT, 256, 0, stream>>>(valB, ln_g, ln_b, keyb);

  // 8) output projection + residual: out = x + normed @ Wo^T + bo
  wmma_gemm_bias<<<gemmGrid, 256, 0, stream>>>(keyb, Wob, bo, x, out, MTOT, DIMD, DIMD);
}
